// LargeSparseResNet_9912784519366
// MI455X (gfx1250) — matrix operand order repaired and checked by reading, not compiled, not run on hardware
//
#include <hip/hip_runtime.h>
#include <hip/hip_bf16.h>

// ---------------------------------------------------------------------------
// CDNA5 (gfx1250, wave32) implicit-GEMM 3D conv ResNet.
//   conv3d_wmma_kernel<FULLC> : D = A(im2col f16) x B(weights f16) via
//     v_wmma_f32_16x16x32_f16, f32 epilogue (BN+res+ReLU).
//   K-loop = kernel taps x 32-channel chunks with register-staged LDS
//   double buffering (1 barrier/chunk, global loads overlap the WMMA).
//   FULLC=true  -> Ci % 32 == 0 : single padding mask per tap, unmasked B.
//   FULLC=false -> Ci < 32 (stage 1): per-element channel masks.
// ---------------------------------------------------------------------------

typedef __attribute__((ext_vector_type(16))) _Float16 v16h;
typedef __attribute__((ext_vector_type(8)))  float    v8f;

#define WAVES 8
#define BLOCK (WAVES * 32)

struct ConvP {
    int N, Di, Hi, Wi, Ci;       // input dims (NDHWC)
    int Do_, Ho, Wo, Co;         // output dims
    int kd, kh, kw;              // kernel
    int sd, sh, sw;              // stride
    int pd, ph, pw;              // pad
    int Mtot;                    // N*Do*Ho*Wo   (GEMM M)
    int relu;
    int addres;                  // add residual before ReLU
};

union Frag { v16h v; unsigned int u[8]; };
union H4   { _Float16 h[4]; uint2 u; };

template <bool FULLC>
__global__ __launch_bounds__(BLOCK)
void conv3d_wmma_kernel(const float* __restrict__ in,
                        const float* __restrict__ wgt,   // [kd,kh,kw,Ci,Co]
                        const float* __restrict__ gamma,
                        const float* __restrict__ beta,
                        const float* __restrict__ res,
                        float* __restrict__ out,
                        ConvP p)
{
    // Double-buffered tiles.  A: 16x32 f16 per wave.  B: 32(K)x16(N) shared,
    // stored [n][k] so ISA K-pairs are contiguous (merges to ds_load_b128).
    __shared__ __align__(16) _Float16 sA[2][WAVES][16][32];
    __shared__ __align__(16) _Float16 sB[2][16][32];

    const int tid  = threadIdx.x;
    const int lane = tid & 31;
    const int wv   = tid >> 5;

    const int m0 = (blockIdx.x * WAVES + wv) * 16;   // 16 output voxels
    const int n0 = blockIdx.y * 16;                  // 16 output channels

    // ---- decode this lane's output voxel (row = lane&15) ONCE
    const int  mrow = lane & 15;
    const int  gm   = m0 + mrow;
    const bool mval = gm < p.Mtot;
    int t = mval ? gm : 0;
    const int ow = t % p.Wo;  t /= p.Wo;
    const int oh = t % p.Ho;  t /= p.Ho;
    const int od = t % p.Do_;
    const int nb = t / p.Do_;
    const int iw0 = ow * p.sw - p.pw;
    const int ih0 = oh * p.sh - p.ph;
    const int id0 = od * p.sd - p.pd;

    const int chalf = (lane >> 4) << 4;   // 0 / 16 : this lane's channel half
    const int ntaps = p.kd * p.kh * p.kw;
    const int nch   = ntaps * ((p.Ci + 31) >> 5);   // total K chunks

    // ---- "next tile to load" state (tap coords kept incrementally)
    int    ltw = 0, lth = 0, ltd = 0, lci = 0;
    size_t lwbase = 0;
    bool   lav;
    size_t labase;
    float  lamask;
    {
        const int iw = iw0, ih = ih0, id = id0;
        lav = mval && ((unsigned)iw < (unsigned)p.Wi) &&
                      ((unsigned)ih < (unsigned)p.Hi) &&
                      ((unsigned)id < (unsigned)p.Di);
        labase = lav ? ((((size_t)nb * p.Di + id) * p.Hi + ih) * p.Wi + iw)
                         * p.Ci : 0;
        lamask = lav ? 1.f : 0.f;
    }
    auto advance = [&]() {
        lci += 32;
        if (lci >= p.Ci) {
            lci = 0;
            lwbase += (size_t)p.Ci * p.Co;
            if (++ltw == p.kw) { ltw = 0; if (++lth == p.kh) { lth = 0; ++ltd; } }
            const int iw = iw0 + ltw, ih = ih0 + lth, id = id0 + ltd;
            lav = mval && ((unsigned)iw < (unsigned)p.Wi) &&
                          ((unsigned)ih < (unsigned)p.Hi) &&
                          ((unsigned)id < (unsigned)p.Di);
            labase = lav ? ((((size_t)nb * p.Di + id) * p.Hi + ih) * p.Wi + iw)
                             * p.Ci : 0;
            lamask = lav ? 1.f : 0.f;
        }
    };

    // ---- prologue: fill tile 0 directly
    {
#pragma unroll
        for (int q = 0; q < 4; ++q) {
            const int cc = lci + chalf + q * 4;
            float msk; size_t idx;
            if (FULLC) { msk = lamask; idx = labase + cc; }
            else {
                const bool ok = lav && (cc < p.Ci);
                msk = ok ? 1.f : 0.f; idx = ok ? labase + cc : 0;
            }
            const float4 f = *reinterpret_cast<const float4*>(&in[idx]);
            H4 pk;
            pk.h[0] = (_Float16)(f.x * msk);
            pk.h[1] = (_Float16)(f.y * msk);
            pk.h[2] = (_Float16)(f.z * msk);
            pk.h[3] = (_Float16)(f.w * msk);
            *reinterpret_cast<uint2*>(&sA[0][wv][mrow][chalf + q * 4]) = pk.u;
        }
        if (tid < 128) {
            const int k  = tid >> 2;
            const int n4 = (tid & 3) << 2;
            float msk; size_t idx;
            if (FULLC) { msk = 1.f; idx = lwbase + (size_t)(lci + k) * p.Co + n0 + n4; }
            else {
                const bool ok = (lci + k) < p.Ci;
                msk = ok ? 1.f : 0.f;
                idx = ok ? lwbase + (size_t)(lci + k) * p.Co + n0 + n4 : 0;
            }
            const float4 f = *reinterpret_cast<const float4*>(&wgt[idx]);
            sB[0][n4 + 0][k] = (_Float16)(f.x * msk);
            sB[0][n4 + 1][k] = (_Float16)(f.y * msk);
            sB[0][n4 + 2][k] = (_Float16)(f.z * msk);
            sB[0][n4 + 3][k] = (_Float16)(f.w * msk);
        }
    }
    advance();

    v8f acc = {};
    int buf = 0;

    for (int ch = 0; ch < nch; ++ch) {
        __syncthreads();                     // tile[buf] ready for all waves
        const bool more = (ch + 1) < nch;

        // ---- issue next chunk's global loads into registers (no wait yet)
        float4 ra[4]; float ma[4];
        float4 rb;    float mb = 1.f;
        if (more) {
#pragma unroll
            for (int q = 0; q < 4; ++q) {
                const int cc = lci + chalf + q * 4;
                size_t idx;
                if (FULLC) { ma[q] = lamask; idx = labase + cc; }
                else {
                    const bool ok = lav && (cc < p.Ci);
                    ma[q] = ok ? 1.f : 0.f; idx = ok ? labase + cc : 0;
                }
                ra[q] = *reinterpret_cast<const float4*>(&in[idx]);
            }
            if (tid < 128) {
                const int k  = tid >> 2;
                const int n4 = (tid & 3) << 2;
                size_t idx;
                if (FULLC) { idx = lwbase + (size_t)(lci + k) * p.Co + n0 + n4; }
                else {
                    const bool ok = (lci + k) < p.Ci;
                    mb  = ok ? 1.f : 0.f;
                    idx = ok ? lwbase + (size_t)(lci + k) * p.Co + n0 + n4 : 0;
                }
                rb = *reinterpret_cast<const float4*>(&wgt[idx]);
            }
        }

        // ---- gather fragments per CDNA5 16-bit A/B VGPR layout (ISA 7.12.2)
        Frag fa, fb;
        const int half = (lane & 16) ? 1 : 0;
        const int akb  = half ? 8 : 0;       // A: upper lanes hold K+8 pairs
#pragma unroll
        for (int j = 0; j < 8; ++j) {
            int ak = ((j < 4) ? (2 * j) : (16 + 2 * (j - 4))) + akb;
            fa.u[j] = *reinterpret_cast<const unsigned int*>(
                          &sA[buf][wv][mrow][ak]);
        }
        const int bkb = half ? 8 : 0;        // B: upper lanes hold K+8 pairs (same as A)
#pragma unroll
        for (int j = 0; j < 8; ++j) {
            fb.u[j] = *reinterpret_cast<const unsigned int*>(
                          &sB[buf][lane & 15][((j < 4) ? (2 * j) : (16 + 2 * (j - 4))) + bkb]);
        }

        acc = __builtin_amdgcn_wmma_f32_16x16x32_f16(
            /*neg_a=*/false, fa.v, /*neg_b=*/false, fb.v,
            /*c_mod=*/(short)0, acc, /*reuse_a=*/false, /*reuse_b=*/false);

        // ---- commit staged registers into the other LDS buffer
        if (more) {
            const int nb2 = buf ^ 1;
#pragma unroll
            for (int q = 0; q < 4; ++q) {
                H4 pk;
                pk.h[0] = (_Float16)(ra[q].x * ma[q]);
                pk.h[1] = (_Float16)(ra[q].y * ma[q]);
                pk.h[2] = (_Float16)(ra[q].z * ma[q]);
                pk.h[3] = (_Float16)(ra[q].w * ma[q]);
                *reinterpret_cast<uint2*>(
                    &sA[nb2][wv][mrow][chalf + q * 4]) = pk.u;
            }
            if (tid < 128) {
                const int k  = tid >> 2;
                const int n4 = (tid & 3) << 2;
                sB[nb2][n4 + 0][k] = (_Float16)(rb.x * mb);
                sB[nb2][n4 + 1][k] = (_Float16)(rb.y * mb);
                sB[nb2][n4 + 2][k] = (_Float16)(rb.z * mb);
                sB[nb2][n4 + 3][k] = (_Float16)(rb.w * mb);
            }
            advance();
        }
        buf ^= 1;
    }

    // ---- f32 epilogue: folded BN, optional residual, ReLU.
    // Co is always a multiple of 16 here -> no N bounds check.
    const int nc = n0 + (lane & 15);
    {
        const float sc = gamma[nc];
        const float bs = beta[nc];
#pragma unroll
        for (int r = 0; r < 8; ++r) {
            const int gmo = m0 + r + ((lane & 16) ? 8 : 0);
            if (gmo < p.Mtot) {
                float v = acc[r] * sc + bs;
                if (p.addres) v += res[(size_t)gmo * p.Co + nc];
                if (p.relu)   v = v > 0.f ? v : 0.f;
                out[(size_t)gmo * p.Co + nc] = v;
            }
        }
    }
}

// Global mean+max pooling; h[b, 0:C]=mean, h[b, C:2C]=max (ref's swapped names)
__global__ void pool_kernel(const float* __restrict__ feat,
                            float* __restrict__ h, int V, int C)
{
    int b = blockIdx.x;
    int c = blockIdx.y * blockDim.x + threadIdx.x;
    if (c >= C) return;
    const float* f = feat + (size_t)b * V * C + c;
    float s = 0.f, mx = -3.402823466e38f;
    for (int v = 0; v < V; ++v) {
        float x = f[(size_t)v * C];
        s += x;
        mx = x > mx ? x : mx;
    }
    h[b * 2 * C + c]     = s / (float)V;
    h[b * 2 * C + C + c] = mx;
}

__global__ void fc_relu_kernel(const float* __restrict__ x,
                               const float* __restrict__ w,   // [Din, Dout]
                               const float* __restrict__ bias,
                               float* __restrict__ y,
                               int B, int Din, int Dout)
{
    int i = blockIdx.x * blockDim.x + threadIdx.x;
    if (i >= B * Dout) return;
    int b = i / Dout, o = i % Dout;
    const float* xr = x + (size_t)b * Din;
    float s = bias[o];
    for (int d = 0; d < Din; ++d) s += xr[d] * w[(size_t)d * Dout + o];
    y[i] = s > 0.f ? s : 0.f;
}

// ---------------------------------------------------------------------------

struct StageCfg { int n, cout, kd, kh, kw, sd, sh, sw; };
static const StageCfg STAGE_CFG[7] = {
    {3,  16, 3, 3, 3, 1, 1, 1},
    {3,  32, 3, 3, 3, 2, 2, 2},
    {3,  64, 3, 3, 3, 1, 1, 1},
    {3, 128, 3, 3, 3, 3, 3, 3},
    {3, 256, 3, 3, 3, 1, 1, 1},
    {3, 512, 3, 3, 3, 1, 1, 1},
    {1, 512, 1, 3, 1, 1, 2, 1},
};

static void launch_conv(const float* in, const float* W, const float* g,
                        const float* b, const float* res, float* out,
                        const ConvP& p, hipStream_t stream)
{
    int mtiles = (p.Mtot + 15) / 16;
    dim3 grid((mtiles + WAVES - 1) / WAVES, p.Co / 16);
    if (p.Ci % 32 == 0) {
        conv3d_wmma_kernel<true><<<grid, BLOCK, 0, stream>>>(
            in, W, g, b, res ? res : out, out, p);
    } else {
        conv3d_wmma_kernel<false><<<grid, BLOCK, 0, stream>>>(
            in, W, g, b, res ? res : out, out, p);
    }
}

extern "C" void kernel_launch(void* const* d_in, const int* in_sizes, int n_in,
                              void* d_out, int out_size, void* d_ws, size_t ws_size,
                              hipStream_t stream)
{
    (void)in_sizes; (void)n_in; (void)out_size; (void)ws_size;

    const float* x = (const float*)d_in[0];
    int cursor = 1;

    // Workspace: three ping-pong activation buffers (largest = 4*64^3*16 f32
    // = 64 MiB) + pooled vector + MLP intermediate.
    const size_t bufElems = (size_t)4 * 64 * 64 * 64 * 16;
    float* bufA = (float*)d_ws;
    float* bufB = bufA + bufElems;
    float* bufC = bufB + bufElems;
    float* hbuf = bufC + bufElems;     // [4,1024]
    float* tmp1 = hbuf + 4 * 1024;     // [4,256]

    float* bufs[3] = {bufA, bufB, bufC};
    auto pick = [&](const float* a, const float* b2) -> float* {
        for (int i = 0; i < 3; ++i)
            if (bufs[i] != a && bufs[i] != b2) return bufs[i];
        return bufs[0];
    };

    int N = 4, Di = 64, Hi = 64, Wi = 64, Ci = 4;
    const float* cur = x;

    for (int s = 0; s < 7; ++s) {
        const StageCfg& c = STAGE_CFG[s];

        ConvP p{};
        p.N = N; p.Di = Di; p.Hi = Hi; p.Wi = Wi; p.Ci = Ci;
        p.kd = c.kd; p.kh = c.kh; p.kw = c.kw;
        p.sd = c.sd; p.sh = c.sh; p.sw = c.sw;
        p.pd = c.kd / 2; p.ph = c.kh / 2; p.pw = c.kw / 2;
        p.Co  = c.cout;
        p.Do_ = (Di + 2 * p.pd - c.kd) / c.sd + 1;
        p.Ho  = (Hi + 2 * p.ph - c.kh) / c.sh + 1;
        p.Wo  = (Wi + 2 * p.pw - c.kw) / c.sw + 1;
        p.Mtot = N * p.Do_ * p.Ho * p.Wo;
        p.relu = 1; p.addres = 0;

        // SparseConvBlock: conv + folded-BN + ReLU
        const float* W = (const float*)d_in[cursor++];
        const float* g = (const float*)d_in[cursor++];
        const float* b = (const float*)d_in[cursor++];
        float* dst = pick(cur, nullptr);
        launch_conv(cur, W, g, b, nullptr, dst, p, stream);
        cur = dst;

        Di = p.Do_; Hi = p.Ho; Wi = p.Wo; Ci = c.cout;

        // SparseResBlocks: stride-1, Cin==Cout, identity shortcut
        ConvP q = p;
        q.Di = Di; q.Hi = Hi; q.Wi = Wi; q.Ci = Ci;
        q.sd = q.sh = q.sw = 1;
        q.Do_ = Di; q.Ho = Hi; q.Wo = Wi;
        q.Mtot = N * Di * Hi * Wi;

        for (int r = 0; r < c.n - 1; ++r) {
            const float* w1 = (const float*)d_in[cursor++];
            const float* g1 = (const float*)d_in[cursor++];
            const float* b1 = (const float*)d_in[cursor++];
            const float* w2 = (const float*)d_in[cursor++];
            const float* g2 = (const float*)d_in[cursor++];
            const float* b2 = (const float*)d_in[cursor++];

            float* y = pick(cur, nullptr);
            q.relu = 1; q.addres = 0;
            launch_conv(cur, w1, g1, b1, nullptr, y, q, stream);

            float* z = pick(cur, y);
            q.relu = 1; q.addres = 1;          // relu(x + y)
            launch_conv(y, w2, g2, b2, cur, z, q, stream);
            cur = z;
        }
    }

    // Global pooling: mean (first 512) + max (second 512) -> [4,1024]
    int V = Di * Hi * Wi;   // 11*6*11 = 726
    dim3 pg(4, (512 + 255) / 256);
    pool_kernel<<<pg, 256, 0, stream>>>(cur, hbuf, V, 512);

    // MLP 1024 -> 256 -> 128, ReLU
    const float* mw1 = (const float*)d_in[cursor++];
    const float* mb1 = (const float*)d_in[cursor++];
    const float* mw2 = (const float*)d_in[cursor++];
    const float* mb2 = (const float*)d_in[cursor++];
    fc_relu_kernel<<<(4 * 256 + 255) / 256, 256, 0, stream>>>(
        hbuf, mw1, mb1, tmp1, 4, 1024, 256);
    fc_relu_kernel<<<(4 * 128 + 255) / 256, 256, 0, stream>>>(
        tmp1, mw2, mb2, (float*)d_out, 4, 256, 128);
}
